// build_model_12945031430275
// MI455X (gfx1250) — matrix operand order repaired and checked by reading, not compiled, not run on hardware
//
#include <hip/hip_runtime.h>
#include <hip/hip_bf16.h>
#include <math.h>

// ---------------------------------------------------------------------------
// GRU world-model forward, CDNA5 (gfx1250), bf16 WMMA w/ fp32 accumulate.
// B=256, TH=32, TF=16, S=512, A=1, F=1024, L=2, prefix T=65.
// Batched (input-side) GEMMs: 32x32 tile/wave (4 WMMA per 8 b128 loads).
// Sequential GRU step: 16x16 tile/wave (max parallelism on critical path).
// ---------------------------------------------------------------------------

typedef unsigned short ushort_t;
typedef __attribute__((ext_vector_type(16))) __bf16        v16bf;
typedef __attribute__((ext_vector_type(8)))  float         v8f;
typedef __attribute__((ext_vector_type(4)))  unsigned int  uint4v;

#define Bc    256
#define THc   32
#define TFc   16
#define Sc    512
#define Fc    1024
#define G3F   3072         // 3*F
#define Tpre  65           // 2*TH + 1

union FragBF { v16bf v; uint4v q[2]; };

__device__ __forceinline__ unsigned short f2bf(float x) {
  unsigned u = __float_as_uint(x);
  u += 0x7fffu + ((u >> 16) & 1u);          // round to nearest even
  return (unsigned short)(u >> 16);
}
__device__ __forceinline__ float bf2f(unsigned short h) {
  return __uint_as_float(((unsigned)h) << 16);
}
__device__ __forceinline__ float sigmoidf_(float x) {
  return 1.0f / (1.0f + __expf(-x));
}

__device__ __forceinline__ void load_afrag(FragBF& f, const ushort_t* row, int k0, int half) {
  // A operand: lane=M; halves hold K 0-7/8-15 then 16-23/24-31.
  f.q[0] = *(const uint4v*)(row + k0 + half * 8);
  f.q[1] = *(const uint4v*)(row + k0 + 16 + half * 8);
}
__device__ __forceinline__ void load_bfrag(FragBF& f, const ushort_t* row, int k0, int half) {
  // B operand: lane=N; halves hold K 0-7/8-15 then 16-23/24-31.
  f.q[0] = *(const uint4v*)(row + k0 + half * 8);
  f.q[1] = *(const uint4v*)(row + k0 + 16 + half * 8);
}
#define WMMA_BF16(acc, a, b) \
  acc = __builtin_amdgcn_wmma_f32_16x16x32_bf16(false, (a).v, false, (b).v, \
                                                (short)0, acc, false, false)

// ---------------------------------------------------------------------------
// Generic bf16 WMMA GEMM:  C[M,N] = act( A[M,K] @ W[N,K]^T + bias[N] )
// One wave computes a 32x32 (MxN) tile: 2 A-frags + 2 B-frags -> 4 WMMAs/chunk.
// Optional fp32 output (ldcf) and/or bf16 output (ldcb).
// Grid: (N/32, M/128), block 128 (4 waves).
// ---------------------------------------------------------------------------
__global__ __launch_bounds__(128)
void gemm_bf16_kernel(const ushort_t* __restrict__ A,
                      const ushort_t* __restrict__ W,
                      const float*    __restrict__ bias,
                      float* __restrict__ Cf, int ldcf,
                      ushort_t* __restrict__ Cb, int ldcb,
                      int K, int act)
{
  const int lane = threadIdx.x & 31;
  const int wid  = threadIdx.x >> 5;
  const int half = lane >> 4;
  const int r    = lane & 15;

  const int mrow = (blockIdx.y * 4 + wid) * 32;   // A row base (M tile, 32 rows)
  const int n0   = blockIdx.x * 32;               // output col base (N tile, 32 cols)

  const ushort_t* arow0 = A + (size_t)(mrow + r) * K;
  const ushort_t* arow1 = A + (size_t)(mrow + 16 + r) * K;
  const ushort_t* w0    = W + (size_t)(n0 + r) * K;
  const ushort_t* w1    = W + (size_t)(n0 + 16 + r) * K;

  v8f acc00 = {}, acc01 = {}, acc10 = {}, acc11 = {};
  for (int k0 = 0; k0 < K; k0 += 32) {
    __builtin_prefetch((const void*)(w0 + k0 + 128), 0, 0);
    __builtin_prefetch((const void*)(w1 + k0 + 128), 0, 0);
    FragBF a0, a1, b0, b1;
    load_afrag(a0, arow0, k0, half);
    load_afrag(a1, arow1, k0, half);
    load_bfrag(b0, w0, k0, half);
    load_bfrag(b1, w1, k0, half);
    WMMA_BF16(acc00, a0, b0);
    WMMA_BF16(acc01, a0, b1);
    WMMA_BF16(acc10, a1, b0);
    WMMA_BF16(acc11, a1, b1);
  }

  const int c0 = n0 + r, c1 = n0 + 16 + r;
  const float bb0 = bias ? bias[c0] : 0.f;
  const float bb1 = bias ? bias[c1] : 0.f;
#pragma unroll
  for (int v = 0; v < 8; ++v) {
    const int row0 = mrow + half * 8 + v;           // D: lane=N, vgpr=M(+8*half)
    const int row1 = row0 + 16;
    float x00 = acc00[v] + bb0, x01 = acc01[v] + bb1;
    float x10 = acc10[v] + bb0, x11 = acc11[v] + bb1;
    if (act == 1) {
      x00 = tanhf(x00); x01 = tanhf(x01);
      x10 = tanhf(x10); x11 = tanhf(x11);
    }
    if (Cf) {
      Cf[(size_t)row0 * ldcf + c0] = x00;  Cf[(size_t)row0 * ldcf + c1] = x01;
      Cf[(size_t)row1 * ldcf + c0] = x10;  Cf[(size_t)row1 * ldcf + c1] = x11;
    }
    if (Cb) {
      Cb[(size_t)row0 * ldcb + c0] = f2bf(x00);  Cb[(size_t)row0 * ldcb + c1] = f2bf(x01);
      Cb[(size_t)row1 * ldcb + c0] = f2bf(x10);  Cb[(size_t)row1 * ldcb + c1] = f2bf(x11);
    }
  }
}

// ---------------------------------------------------------------------------
// Fused GRU step: gh = h_in @ Whh^T (+bhh), combine with precomputed
// gi = x @ Wih^T + bih (bf16), apply gates, write h_out (fp32 + bf16 mirror,
// optional bf16 sequence slot). Hidden state is double-buffered by the host
// to avoid intra-grid read/write races.
// Grid: (F/16, B/64), block 128 (4 waves); one wave = 16(batch)x16(feat) tile
// with 3 gate accumulators -> kept small to maximize wave parallelism on the
// latency-critical sequential chain.
// ---------------------------------------------------------------------------
__global__ __launch_bounds__(128)
void gru_step_kernel(const ushort_t* __restrict__ Gt,    // [B, 3F] bf16 (gi+bih)
                     const ushort_t* __restrict__ Whh,   // [3F, F] bf16
                     const float*    __restrict__ bhh,   // [3F]
                     const float*    __restrict__ hf_in, // [B, F] fp32
                     const ushort_t* __restrict__ hb_in, // [B, F] bf16
                     float*    __restrict__ hf_out,
                     ushort_t* __restrict__ hb_out,
                     ushort_t* __restrict__ seq_out)     // optional [B, F]
{
  const int lane = threadIdx.x & 31;
  const int wid  = threadIdx.x >> 5;
  const int half = lane >> 4;
  const int r    = lane & 15;

  const int mrow = (blockIdx.y * 4 + wid) * 16;   // batch tile
  const int fcol = blockIdx.x * 16;               // feature tile

  const ushort_t* arow = hb_in + (size_t)(mrow + r) * Fc;
  const ushort_t* wr   = Whh + (size_t)(0 * Fc + fcol + r) * Fc;
  const ushort_t* wz   = Whh + (size_t)(1 * Fc + fcol + r) * Fc;
  const ushort_t* wn   = Whh + (size_t)(2 * Fc + fcol + r) * Fc;

  v8f accR = {}, accZ = {}, accN = {};
  for (int k0 = 0; k0 < Fc; k0 += 32) {
    __builtin_prefetch((const void*)(wr + k0 + 128), 0, 0);
    FragBF a, br_, bz_, bn_;
    load_afrag(a, arow, k0, half);
    load_bfrag(br_, wr, k0, half);
    load_bfrag(bz_, wz, k0, half);
    load_bfrag(bn_, wn, k0, half);
    WMMA_BF16(accR, a, br_);
    WMMA_BF16(accZ, a, bz_);
    WMMA_BF16(accN, a, bn_);
  }

  const int   f  = fcol + r;
  const float bR = bhh[f];
  const float bZ = bhh[Fc + f];
  const float bN = bhh[2 * Fc + f];
#pragma unroll
  for (int v = 0; v < 8; ++v) {
    const int b = mrow + half * 8 + v;
    const size_t gbase = (size_t)b * G3F;
    const float gi_r = bf2f(Gt[gbase + f]);
    const float gi_z = bf2f(Gt[gbase + Fc + f]);
    const float gi_n = bf2f(Gt[gbase + 2 * Fc + f]);
    const float rg = sigmoidf_(gi_r + accR[v] + bR);
    const float zg = sigmoidf_(gi_z + accZ[v] + bZ);
    const float ng = tanhf(gi_n + rg * (accN[v] + bN));
    const float ho = hf_in[(size_t)b * Fc + f];
    const float hn = (1.f - zg) * ng + zg * ho;
    hf_out[(size_t)b * Fc + f] = hn;
    const unsigned short hnb = f2bf(hn);
    hb_out[(size_t)b * Fc + f] = hnb;
    if (seq_out) seq_out[(size_t)b * Fc + f] = hnb;
  }
}

// ---------------------------------------------------------------------------
// Small helper kernels
// ---------------------------------------------------------------------------
__global__ void cvt_f32_bf16_kernel(const float* __restrict__ in,
                                    ushort_t* __restrict__ out, int n) {
  int i = blockIdx.x * blockDim.x + threadIdx.x;
  if (i < n) out[i] = f2bf(in[i]);
}

// Scatter state embeddings [B*TH, F] -> X[t=2*th][b][f]
__global__ void interleave_semb_kernel(const ushort_t* __restrict__ semb,
                                       ushort_t* __restrict__ X) {
  int i = blockIdx.x * blockDim.x + threadIdx.x;   // over B*TH*F
  int f = i & (Fc - 1);
  int row = i >> 10;          // b*TH + th
  int th = row & (THc - 1);
  int b  = row >> 5;
  X[((size_t)(2 * th) * Bc + b) * Fc + f] = semb[i];
}

// X[t=2*th+1][b][f] = bf16( tanh( history_a[b,th] * Wa[f] + ba[f] ) )  (A==1)
__global__ void embed_action_hist_kernel(const float* __restrict__ ha,
                                         const float* __restrict__ Wa,
                                         const float* __restrict__ ba,
                                         ushort_t* __restrict__ X) {
  int i = blockIdx.x * blockDim.x + threadIdx.x;   // over B*TH*F
  int f = i & (Fc - 1);
  int row = i >> 10;
  int th = row & (THc - 1);
  int b  = row >> 5;
  float v = tanhf(ha[row] * Wa[f] + ba[f]);
  X[((size_t)(2 * th + 1) * Bc + b) * Fc + f] = f2bf(v);
}

// out[b][f] = bf16( tanh( future_a[b, t] * Wa[f] + ba[f] ) )
__global__ void embed_action_fut_kernel(const float* __restrict__ fa, int t,
                                        const float* __restrict__ Wa,
                                        const float* __restrict__ ba,
                                        ushort_t* __restrict__ out) {
  int i = blockIdx.x * blockDim.x + threadIdx.x;   // over B*F
  int f = i & (Fc - 1);
  int b = i >> 10;
  out[i] = f2bf(tanhf(fa[b * TFc + t] * Wa[f] + ba[f]));
}

// reward[b, t] = sigmoid( dot(h1[b,:], Wr) + br )   (R==1)
__global__ void reward_kernel(const float* __restrict__ h1,
                              const float* __restrict__ Wr,
                              const float* __restrict__ br,
                              float* __restrict__ out_r, int t) {
  int b = threadIdx.x;   // 256 threads
  float s = 0.f;
  for (int j = 0; j < Fc; ++j) s += h1[(size_t)b * Fc + j] * Wr[j];
  out_r[b * TFc + t] = 1.0f / (1.0f + __expf(-(s + br[0])));
}

// ---------------------------------------------------------------------------
// Host orchestration
// ---------------------------------------------------------------------------
extern "C" void kernel_launch(void* const* d_in, const int* in_sizes, int n_in,
                              void* d_out, int out_size, void* d_ws, size_t ws_size,
                              hipStream_t stream)
{
  (void)in_sizes; (void)n_in; (void)out_size; (void)ws_size;

  const float* history_s = (const float*)d_in[0];
  const float* history_a = (const float*)d_in[1];
  const float* present_s = (const float*)d_in[2];
  const float* future_a  = (const float*)d_in[3];
  const float* Ws  = (const float*)d_in[4];
  const float* bs  = (const float*)d_in[5];
  const float* Wa  = (const float*)d_in[6];
  const float* ba  = (const float*)d_in[7];
  const float* Wih = (const float*)d_in[8];
  const float* Whh = (const float*)d_in[9];
  const float* bih = (const float*)d_in[10];
  const float* bhh = (const float*)d_in[11];
  const float* Wr  = (const float*)d_in[12];
  const float* br  = (const float*)d_in[13];
  const float* Ws2 = (const float*)d_in[14];
  const float* bs2 = (const float*)d_in[15];

  float* out_r = (float*)d_out;                 // [B, TF, 1]
  float* out_s = out_r + Bc * TFc;              // [B, TF, S]

  // ---- workspace carving (256B aligned) ----
  char* p = (char*)d_ws;
  auto carve = [&](size_t bytes) -> char* {
    char* q = p; p += (bytes + 255) & ~(size_t)255; return q;
  };
  const size_t LW = (size_t)G3F * Fc;           // per-layer weight elements

  ushort_t* Ws_b  = (ushort_t*)carve((size_t)Fc * Sc * 2);
  ushort_t* Wih_b = (ushort_t*)carve(2 * LW * 2);
  ushort_t* Whh_b = (ushort_t*)carve(2 * LW * 2);
  ushort_t* Ws2_b = (ushort_t*)carve((size_t)Sc * Fc * 2);
  ushort_t* hs_b  = (ushort_t*)carve((size_t)Bc * THc * Sc * 2);
  ushort_t* ps_b  = (ushort_t*)carve((size_t)Bc * Sc * 2);
  ushort_t* semb  = (ushort_t*)carve((size_t)Bc * THc * Fc * 2);
  ushort_t* X     = (ushort_t*)carve((size_t)Tpre * Bc * Fc * 2);
  ushort_t* H0seq = (ushort_t*)carve((size_t)Tpre * Bc * Fc * 2);
  ushort_t* G     = (ushort_t*)carve((size_t)Tpre * Bc * G3F * 2);  // also Gi scratch
  float*    h0f[2]; ushort_t* h0b[2]; float* h1f[2]; ushort_t* h1b[2];
  h0f[0] = (float*)carve((size_t)Bc * Fc * 4);
  h0f[1] = (float*)carve((size_t)Bc * Fc * 4);
  h1f[0] = (float*)carve((size_t)Bc * Fc * 4);
  h1f[1] = (float*)carve((size_t)Bc * Fc * 4);
  h0b[0] = (ushort_t*)carve((size_t)Bc * Fc * 2);
  h0b[1] = (ushort_t*)carve((size_t)Bc * Fc * 2);
  h1b[0] = (ushort_t*)carve((size_t)Bc * Fc * 2);
  h1b[1] = (ushort_t*)carve((size_t)Bc * Fc * 2);
  ushort_t* xf_b = (ushort_t*)carve((size_t)Bc * Fc * 2);
  ushort_t* s_b  = (ushort_t*)carve((size_t)Bc * Sc * 2);

  auto cvt = [&](const float* in, ushort_t* out, size_t n) {
    cvt_f32_bf16_kernel<<<(unsigned)((n + 255) / 256), 256, 0, stream>>>(in, out, (int)n);
  };
  auto gemm = [&](const ushort_t* Am, const ushort_t* Wm, const float* bias,
                  float* Cf, int ldcf, ushort_t* Cb, int ldcb,
                  int M, int N, int K, int act) {
    dim3 grid(N / 32, M / 128);
    gemm_bf16_kernel<<<grid, 128, 0, stream>>>(Am, Wm, bias, Cf, ldcf, Cb, ldcb, K, act);
  };
  auto gru = [&](const ushort_t* Gt, const ushort_t* Whhl, const float* bhhl,
                 float* hfi, ushort_t* hbi, float* hfo, ushort_t* hbo,
                 ushort_t* seq) {
    gru_step_kernel<<<dim3(Fc / 16, Bc / 64), 128, 0, stream>>>(
        Gt, Whhl, bhhl, hfi, hbi, hfo, hbo, seq);
  };

  // ---- Phase 0: bf16 conversions ----
  cvt(Ws,        Ws_b,  (size_t)Fc * Sc);
  cvt(Wih,       Wih_b, 2 * LW);
  cvt(Whh,       Whh_b, 2 * LW);
  cvt(Ws2,       Ws2_b, (size_t)Sc * Fc);
  cvt(history_s, hs_b,  (size_t)Bc * THc * Sc);
  cvt(present_s, ps_b,  (size_t)Bc * Sc);

  // ---- Phase 1: build input sequence X[t][b][f] (bf16) ----
  gemm(hs_b, Ws_b, bs, nullptr, 0, semb, Fc, Bc * THc, Fc, Sc, 1);          // tanh(s@Ws^T+bs)
  interleave_semb_kernel<<<(Bc * THc * Fc) / 256, 256, 0, stream>>>(semb, X);
  embed_action_hist_kernel<<<(Bc * THc * Fc) / 256, 256, 0, stream>>>(history_a, Wa, ba, X);
  gemm(ps_b, Ws_b, bs, nullptr, 0, X + (size_t)(Tpre - 1) * Bc * Fc, Fc,
       Bc, Fc, Sc, 1);                                                       // present @ t=64

  // ---- Phase 2: layer-0 input gates (batched over all t), then scan ----
  gemm(X, Wih_b, bih, nullptr, 0, G, G3F, Tpre * Bc, G3F, Fc, 0);
  hipMemsetAsync(h0f[0], 0, (size_t)Bc * Fc * 4, stream);
  hipMemsetAsync(h0b[0], 0, (size_t)Bc * Fc * 2, stream);
  hipMemsetAsync(h1f[0], 0, (size_t)Bc * Fc * 4, stream);
  hipMemsetAsync(h1b[0], 0, (size_t)Bc * Fc * 2, stream);

  int c0 = 0, c1 = 0;
  for (int t = 0; t < Tpre; ++t) {
    gru(G + (size_t)t * Bc * G3F, Whh_b, bhh,
        h0f[c0], h0b[c0], h0f[1 - c0], h0b[1 - c0],
        H0seq + (size_t)t * Bc * Fc);
    c0 ^= 1;
  }

  // ---- Phase 3: layer-1 input gates (batched), then scan ----
  gemm(H0seq, Wih_b + LW, bih + G3F, nullptr, 0, G, G3F, Tpre * Bc, G3F, Fc, 0);
  for (int t = 0; t < Tpre; ++t) {
    gru(G + (size_t)t * Bc * G3F, Whh_b + LW, bhh + G3F,
        h1f[c1], h1b[c1], h1f[1 - c1], h1b[1 - c1], nullptr);
    c1 ^= 1;
  }

  // ---- Phase 4: future rollout ----
  ushort_t* Gi = G;   // [B, 3F] scratch (prefix gates already consumed)
  for (int t = 0; t < TFc; ++t) {
    // step_layers(H, tanh(a@Wa^T+ba))
    embed_action_fut_kernel<<<(Bc * Fc) / 256, 256, 0, stream>>>(future_a, t, Wa, ba, xf_b);
    gemm(xf_b, Wih_b, bih, nullptr, 0, Gi, G3F, Bc, G3F, Fc, 0);
    gru(Gi, Whh_b, bhh, h0f[c0], h0b[c0], h0f[1 - c0], h0b[1 - c0], nullptr); c0 ^= 1;
    gemm(h0b[c0], Wih_b + LW, bih + G3F, nullptr, 0, Gi, G3F, Bc, G3F, Fc, 0);
    gru(Gi, Whh_b + LW, bhh + G3F, h1f[c1], h1b[c1], h1f[1 - c1], h1b[1 - c1], nullptr); c1 ^= 1;

    // outputs: reward + predicted state (fp32 straight into d_out slice)
    reward_kernel<<<1, 256, 0, stream>>>(h1f[c1], Wr, br, out_r, t);
    gemm(h1b[c1], Ws2_b, bs2, out_s + t * Sc, TFc * Sc, s_b, Sc, Bc, Sc, Fc, 1);

    // step_layers(H1, tanh(s@Ws^T+bs))
    gemm(s_b, Ws_b, bs, nullptr, 0, xf_b, Fc, Bc, Fc, Sc, 1);
    gemm(xf_b, Wih_b, bih, nullptr, 0, Gi, G3F, Bc, G3F, Fc, 0);
    gru(Gi, Whh_b, bhh, h0f[c0], h0b[c0], h0f[1 - c0], h0b[1 - c0], nullptr); c0 ^= 1;
    gemm(h0b[c0], Wih_b + LW, bih + G3F, nullptr, 0, Gi, G3F, Bc, G3F, Fc, 0);
    gru(Gi, Whh_b + LW, bhh + G3F, h1f[c1], h1b[c1], h1f[1 - c1], h1b[1 - c1], nullptr); c1 ^= 1;
  }
}
